// Int8DynActInt4WeightLinear_21706764714507
// MI455X (gfx1250) — 8-bit matrix operand order repaired and checked by reading, compiled, not run on hardware
//
#include <hip/hip_runtime.h>
#include <stdint.h>

typedef int   v8i __attribute__((ext_vector_type(8)));
typedef float v8f __attribute__((ext_vector_type(8)));
typedef int   vi4 __attribute__((__vector_size__(16)));

#define M_TOT 8192
#define K_TOT 4096
#define N_TOT 4096
#define GSZ   256
#define NGRP  16

#define BM 128
#define BN 64

#ifndef __has_builtin
#define __has_builtin(x) 0
#endif

#if __has_builtin(__builtin_amdgcn_global_load_async_to_lds_b128)
#define USE_ASYNC_LDS 1
#else
#define USE_ASYNC_LDS 0
#endif

#if __has_builtin(__builtin_amdgcn_s_wait_asynccnt)
#define WAIT_ASYNC(n) __builtin_amdgcn_s_wait_asynccnt(n)
#else
#define WAIT_ASYNC(n) asm volatile("s_wait_asynccnt %0" ::"i"(n) : "memory")
#endif

#define GPTR(p) ((vi4*)(p))
#define SPTR(p) ((vi4*)(p))

static __device__ __forceinline__ v8i zero_v8i() {
  v8i z = {0, 0, 0, 0, 0, 0, 0, 0};
  return z;
}

__global__ __launch_bounds__(256) void quant_rows_kernel(
    const float* __restrict__ x, int8_t* __restrict__ q,
    float* __restrict__ qscale, float* __restrict__ qzp) {
  __shared__ float sred[256];
  __shared__ float s_mn, s_mx;
  const int row = blockIdx.x;
  const int t   = threadIdx.x;

  const float4* xr = (const float4*)(x + (size_t)row * K_TOT) + t * 4;
  float xs[16];
#pragma unroll
  for (int j = 0; j < 4; ++j) {
    float4 v = xr[j];
    xs[j * 4 + 0] = v.x; xs[j * 4 + 1] = v.y;
    xs[j * 4 + 2] = v.z; xs[j * 4 + 3] = v.w;
  }
  float mn = 0.0f, mx = 0.0f;
#pragma unroll
  for (int i = 0; i < 16; ++i) { mn = fminf(mn, xs[i]); mx = fmaxf(mx, xs[i]); }

  sred[t] = mn; __syncthreads();
  for (int s = 128; s > 0; s >>= 1) {
    if (t < s) sred[t] = fminf(sred[t], sred[t + s]);
    __syncthreads();
  }
  if (t == 0) s_mn = sred[0];
  __syncthreads();
  sred[t] = mx; __syncthreads();
  for (int s = 128; s > 0; s >>= 1) {
    if (t < s) sred[t] = fmaxf(sred[t], sred[t + s]);
    __syncthreads();
  }
  if (t == 0) s_mx = sred[0];
  __syncthreads();

  const float gmn = s_mn, gmx = s_mx;
  const float eps   = 1.1920928955078125e-07f;
  const float scale = fmaxf((gmx - gmn) / 255.0f, eps);
  float zpf = ((-128.0f + gmn / scale) + (127.0f + gmx / scale) > 0.0f)
                  ? (-128.0f - gmn / scale)
                  : (127.0f - gmx / scale);
  zpf = fminf(fmaxf(rintf(zpf), -128.0f), 127.0f);

  union { int8_t b[16]; int4 v; } pk;
#pragma unroll
  for (int i = 0; i < 16; ++i) {
    float qv = fminf(fmaxf(rintf(xs[i] / scale) + zpf, -128.0f), 127.0f);
    pk.b[i] = (int8_t)qv;
  }
  ((int4*)(q + (size_t)row * K_TOT))[t] = pk.v;
  if (t == 0) { qscale[row] = scale; qzp[row] = zpf; }
}

__global__ __launch_bounds__(256) void weight_prep_kernel(
    const int8_t* __restrict__ w, const float* __restrict__ scales,
    const float* __restrict__ zeros, int8_t* __restrict__ wp,
    float* __restrict__ wcorr) {
  __shared__ int   si[256];
  __shared__ float sf[NGRP];
  const int o = blockIdx.x;
  const int t = threadIdx.x;
  const int g = t >> 4;

  union { int8_t b[16]; int4 v; } in, ov;
  in.v = ((const int4*)(w + (size_t)o * K_TOT))[t];
  const int zi = (int)zeros[o * NGRP + g];
  int psum = 0;
#pragma unroll
  for (int i = 0; i < 16; ++i) {
    int wv = (int)in.b[i] - zi;
    ov.b[i] = (int8_t)wv;
    psum += wv;
  }
  ((int4*)(wp + (size_t)o * K_TOT))[t] = ov.v;

  si[t] = psum; __syncthreads();
  if (t < NGRP) {
    int gsum = 0;
#pragma unroll
    for (int j = 0; j < 16; ++j) gsum += si[t * 16 + j];
    sf[t] = scales[o * NGRP + t] * (float)gsum;
  }
  __syncthreads();
  if (t == 0) {
    float acc = 0.0f;
#pragma unroll
    for (int j = 0; j < NGRP; ++j) acc += sf[j];
    wcorr[o] = acc;
  }
}

static __device__ __forceinline__ void mma_group(
    const int8_t* __restrict__ sAp, const int8_t* __restrict__ sBp,
    int wave, int lh, int l15, v8i acc[4]) {
#pragma unroll
  for (int kc = 0; kc < 4; ++kc) {
    union { uint32_t u[8]; v8i v; } av;
    const int8_t* ap = sAp + (wave * 16 + l15) * GSZ + kc * 64 + lh * 8;
    *(uint2*)&av.u[0] = *(const uint2*)(ap + 0);
    *(uint2*)&av.u[2] = *(const uint2*)(ap + 16);
    *(uint2*)&av.u[4] = *(const uint2*)(ap + 32);
    *(uint2*)&av.u[6] = *(const uint2*)(ap + 48);
#pragma unroll
    for (int nt = 0; nt < 4; ++nt) {
      union { uint32_t u[8]; v8i v; } bv;
      const int8_t* bp = sBp + (nt * 16 + l15) * GSZ + kc * 64 + lh * 8;
      *(uint2*)&bv.u[0] = *(const uint2*)(bp + 0);
      *(uint2*)&bv.u[2] = *(const uint2*)(bp + 16);
      *(uint2*)&bv.u[4] = *(const uint2*)(bp + 32);
      *(uint2*)&bv.u[6] = *(const uint2*)(bp + 48);
      acc[nt] = __builtin_amdgcn_wmma_i32_16x16x64_iu8(
          true, av.v, true, bv.v, acc[nt], false, false);
    }
  }
}

__global__ __launch_bounds__(256) void gemm_kernel(
    const int8_t* __restrict__ q, const int8_t* __restrict__ wp,
    const float* __restrict__ qscale, const float* __restrict__ qzp,
    const float* __restrict__ wcorr, const float* __restrict__ scales,
    float* __restrict__ out) {
#if USE_ASYNC_LDS
  __shared__ int8_t sA[2][BM * GSZ];
  __shared__ int8_t sB[2][BN * GSZ];
#else
  __shared__ int8_t sA1[BM * GSZ];
  __shared__ int8_t sB1[BN * GSZ];
#endif
  __shared__ float  sScale[BN * NGRP];
  __shared__ float  sWcorr[BN];
  __shared__ float  sQs[BM];
  __shared__ float  sQz[BM];

  const int t    = threadIdx.x;
  const int wave = t >> 5;
  const int lane = t & 31;
  const int lh   = lane >> 4;
  const int l15  = lane & 15;
  const int m0 = blockIdx.y * BM;
  const int n0 = blockIdx.x * BN;

  int    lofA[8]; size_t gofA[8];
  int    lofB[4]; size_t gofB[4];
#pragma unroll
  for (int j = 0; j < 8; ++j) {
    int c = t + j * 256, row = c >> 4, cb = c & 15;
    lofA[j] = row * GSZ + cb * 16;
    gofA[j] = (size_t)(m0 + row) * K_TOT + cb * 16;
  }
#pragma unroll
  for (int j = 0; j < 4; ++j) {
    int c = t + j * 256, row = c >> 4, cb = c & 15;
    lofB[j] = row * GSZ + cb * 16;
    gofB[j] = (size_t)(n0 + row) * K_TOT + cb * 16;
  }

  ((float4*)sScale)[t] = ((const float4*)(scales + (size_t)n0 * NGRP))[t];
  if (t < BN) sWcorr[t] = wcorr[n0 + t];
  if (t < BM) { sQs[t] = qscale[m0 + t]; sQz[t] = qzp[m0 + t]; }

  v8i acc[4];
  v8f facc[4];
#pragma unroll
  for (int nt = 0; nt < 4; ++nt) {
    acc[nt] = zero_v8i();
#pragma unroll
    for (int r = 0; r < 8; ++r) facc[nt][r] = 0.0f;
  }

#if USE_ASYNC_LDS
  auto issue_stage = [&](int g, int buf) {
    const size_t ko = (size_t)g * GSZ;
#pragma unroll
    for (int j = 0; j < 8; ++j)
      __builtin_amdgcn_global_load_async_to_lds_b128(
          GPTR(q + gofA[j] + ko), SPTR(&sA[buf][lofA[j]]), 0, 0);
#pragma unroll
    for (int j = 0; j < 4; ++j)
      __builtin_amdgcn_global_load_async_to_lds_b128(
          GPTR(wp + gofB[j] + ko), SPTR(&sB[buf][lofB[j]]), 0, 0);
  };

  issue_stage(0, 0);
  for (int g = 0; g < NGRP; ++g) {
    const int cur = g & 1;
    __syncthreads();
    if (g + 1 < NGRP) {
      issue_stage(g + 1, cur ^ 1);
      WAIT_ASYNC(12);
    } else {
      WAIT_ASYNC(0);
    }
    __syncthreads();

    mma_group(sA[cur], sB[cur], wave, lh, l15, acc);

#pragma unroll
    for (int nt = 0; nt < 4; ++nt) {
      float sg = sScale[(nt * 16 + l15) * NGRP + g];
#pragma unroll
      for (int r = 0; r < 8; ++r) facc[nt][r] += sg * (float)acc[nt][r];
      acc[nt] = zero_v8i();
    }
  }
#else
  int4 pa[8], pb[4];
  {
    const size_t ko = 0;
#pragma unroll
    for (int j = 0; j < 8; ++j) pa[j] = *(const int4*)(q + gofA[j] + ko);
#pragma unroll
    for (int j = 0; j < 4; ++j) pb[j] = *(const int4*)(wp + gofB[j] + ko);
  }
  for (int g = 0; g < NGRP; ++g) {
    __syncthreads();
#pragma unroll
    for (int j = 0; j < 8; ++j) *(int4*)(sA1 + lofA[j]) = pa[j];
#pragma unroll
    for (int j = 0; j < 4; ++j) *(int4*)(sB1 + lofB[j]) = pb[j];
    if (g + 1 < NGRP) {
      const size_t ko = (size_t)(g + 1) * GSZ;
#pragma unroll
      for (int j = 0; j < 8; ++j) pa[j] = *(const int4*)(q + gofA[j] + ko);
#pragma unroll
      for (int j = 0; j < 4; ++j) pb[j] = *(const int4*)(wp + gofB[j] + ko);
    }
    __syncthreads();

    mma_group(sA1, sB1, wave, lh, l15, acc);

#pragma unroll
    for (int nt = 0; nt < 4; ++nt) {
      float sg = sScale[(nt * 16 + l15) * NGRP + g];
#pragma unroll
      for (int r = 0; r < 8; ++r) facc[nt][r] += sg * (float)acc[nt][r];
      acc[nt] = zero_v8i();
    }
  }
#endif

#pragma unroll
  for (int nt = 0; nt < 4; ++nt) {
    const int ncol = nt * 16 + l15;
    const float wc = sWcorr[ncol];
#pragma unroll
    for (int r = 0; r < 8; ++r) {
      const int mloc = wave * 16 + lh * 8 + r;
      const float val = sQs[mloc] * (facc[nt][r] - sQz[mloc] * wc);
      out[(size_t)(m0 + mloc) * N_TOT + (n0 + ncol)] = val;
    }
  }
}

extern "C" void kernel_launch(void* const* d_in, const int* in_sizes, int n_in,
                              void* d_out, int out_size, void* d_ws, size_t ws_size,
                              hipStream_t stream) {
  const float*  x      = (const float*)d_in[0];
  const int8_t* w      = (const int8_t*)d_in[1];
  const float*  scales = (const float*)d_in[2];
  const float*  zeros  = (const float*)d_in[3];
  (void)in_sizes; (void)n_in; (void)out_size; (void)ws_size;
  float* out = (float*)d_out;

  int8_t* qbuf = (int8_t*)d_ws;
  int8_t* wpb  = qbuf + (size_t)M_TOT * K_TOT;
  float*  qsc  = (float*)(wpb + (size_t)N_TOT * K_TOT);
  float*  qzp  = qsc + M_TOT;
  float*  wco  = qzp + M_TOT;

  quant_rows_kernel<<<M_TOT, 256, 0, stream>>>(x, qbuf, qsc, qzp);
  weight_prep_kernel<<<N_TOT, 256, 0, stream>>>(w, scales, zeros, wpb, wco);
  gemm_kernel<<<dim3(N_TOT / BN, M_TOT / BM), 256, 0, stream>>>(
      qbuf, wpb, qsc, qzp, wco, scales, out);
}
